// Block_set_86552180949061
// MI455X (gfx1250) — compile-verified, matrix operand order checked by reading, not run on hardware
//
#include <hip/hip_runtime.h>
#include <hip/hip_bf16.h>

// ---------------------------------------------------------------------------
// CDNA5 (gfx1250) implementation: fp32 in HBM, f16 WMMA (f32 accum) for all
// GEMMs including implicit-GEMM 3x3 conv. wave32, 4 waves / 128-thread block.
// Each wave owns a 32x64 output tile: 2 A fragments x 4 B fragments =
// 8x v_wmma_f32_16x16x32_f16 per k-step, with double-buffered LDS staging of
// the 32x64 weight tile (one barrier per k-step, fully unrolled packed-f16
// staging) + global_prefetch of the W stream two steps ahead.
// ---------------------------------------------------------------------------

typedef __attribute__((ext_vector_type(16))) _Float16 v16h;
typedef __attribute__((ext_vector_type(2)))  _Float16 v2h;
typedef __attribute__((ext_vector_type(8)))  float    v8f;

#define B_ 4
#define C_ 256
#define F_ 64
#define T_ 256
#define SCALE_ 0.17677669529663687f   // 1/sqrt(32)

// ---- fragment loaders (CDNA5 ISA 7.12.2 layouts) --------------------------
// A (16x32 f16): lane = h*16 + m ; values j<8 -> K=h*8+j ; j>=8 -> K=16+h*8+(j-8)
__device__ inline v16h load_a_frag(const float* __restrict__ p, int hh) {
  v16h fa;
  float4 a = *(const float4*)(p + hh * 8);
  float4 b = *(const float4*)(p + hh * 8 + 4);
  float4 c = *(const float4*)(p + 16 + hh * 8);
  float4 d = *(const float4*)(p + 16 + hh * 8 + 4);
  fa[0]=(_Float16)a.x;  fa[1]=(_Float16)a.y;  fa[2]=(_Float16)a.z;  fa[3]=(_Float16)a.w;
  fa[4]=(_Float16)b.x;  fa[5]=(_Float16)b.y;  fa[6]=(_Float16)b.z;  fa[7]=(_Float16)b.w;
  fa[8]=(_Float16)c.x;  fa[9]=(_Float16)c.y;  fa[10]=(_Float16)c.z; fa[11]=(_Float16)c.w;
  fa[12]=(_Float16)d.x; fa[13]=(_Float16)d.y; fa[14]=(_Float16)d.z; fa[15]=(_Float16)d.w;
  return fa;
}
// B (32x16): stored transposed in LDS as [n][32 k]; symmetric K mapping.
__device__ inline v16h load_b_frag(const _Float16* __restrict__ row, int hh) {
  v16h fb;
  const _Float16* p = row + hh * 8;
#pragma unroll
  for (int j = 0; j < 8; ++j) { fb[j] = p[j]; fb[8 + j] = p[16 + j]; }
  return fb;
}

// ---------------------------------------------------------------------------
// GroupNorm (over 8ch x 64x256 per (b,g)) reduction -> mean,invstd.
// Strides parametrize NCHW vs channels-last inputs.
// ---------------------------------------------------------------------------
__global__ void gn_reduce_k(const float* __restrict__ in, float* __restrict__ stats,
                            long sb, long sc, long sf, long st, float eps) {
  int bg = blockIdx.x;           // b*32 + g
  int b = bg >> 5, g = bg & 31;
  __shared__ float r1[256];
  __shared__ float r2[256];
  float s1 = 0.f, s2 = 0.f;
  long baseb = (long)b * sb;
  for (int cl = 0; cl < 8; ++cl) {
    long base = baseb + (long)(g * 8 + cl) * sc;
    for (int s = threadIdx.x; s < F_ * T_; s += 256) {
      int f = s >> 8, t = s & 255;
      float v = in[base + (long)f * sf + (long)t * st];
      s1 += v; s2 += v * v;
    }
  }
  r1[threadIdx.x] = s1; r2[threadIdx.x] = s2;
  __syncthreads();
  for (int off = 128; off > 0; off >>= 1) {
    if (threadIdx.x < off) {
      r1[threadIdx.x] += r1[threadIdx.x + off];
      r2[threadIdx.x] += r2[threadIdx.x + off];
    }
    __syncthreads();
  }
  if (threadIdx.x == 0) {
    const float invn = 1.f / (8.f * F_ * T_);
    float m = r1[0] * invn;
    float var = r2[0] * invn - m * m;
    stats[bg * 2] = m;
    stats[bg * 2 + 1] = rsqrtf(var + eps);
  }
}

// normalize NCHW input -> channels-last [b][f][t][c]
__global__ void gn_apply_nchw_to_cl1(const float* __restrict__ in,
                                     const float* __restrict__ stats,
                                     float* __restrict__ out) {
  int idx = blockIdx.x * 256 + threadIdx.x;
  int c = idx & 255, t = (idx >> 8) & 255, f = (idx >> 16) & 63, b = idx >> 22;
  float v = in[(((long)(b * 256 + c) * 64 + f) << 8) + t];
  int sg = (b * 32 + (c >> 3)) * 2;
  out[idx] = (v - stats[sg]) * stats[sg + 1];
}

// normalize channels-last (cl1 -> cl1)
__global__ void gn_apply_cl1(const float* __restrict__ in,
                             const float* __restrict__ stats,
                             float* __restrict__ out) {
  int idx = blockIdx.x * 256 + threadIdx.x;
  int c = idx & 255, b = idx >> 22;
  int sg = (b * 32 + (c >> 3)) * 2;
  out[idx] = (in[idx] - stats[sg]) * stats[sg + 1];
}

// xf[b][t][f][c] = x_nchw + g_res * norm(h2_cl1)
__global__ void gn_apply_res_to_cl2(const float* __restrict__ x,
                                    const float* __restrict__ h2,
                                    const float* __restrict__ stats,
                                    const float* __restrict__ gres,
                                    float* __restrict__ out) {
  int idx = blockIdx.x * 256 + threadIdx.x;
  int c = idx & 255, f = (idx >> 8) & 63, t = (idx >> 14) & 255, b = idx >> 22;
  float hv = h2[(((long)(b * 64 + f) * 256 + t) << 8) + c];
  float xv = x[(((long)(b * 256 + c) * 64 + f) << 8) + t];
  int sg = (b * 32 + (c >> 3)) * 2;
  out[idx] = xv + gres[0] * (hv - stats[sg]) * stats[sg + 1];
}

// ---------------------------------------------------------------------------
// Generic WMMA GEMM: out[m][n] = sum_k A[row(m)][k] * W[k][n] (+bias)
// row(m): m = (b*D1 + i1)*D2 + i2, base = b*sB + i1*s1 + i2*s2 (elements).
// Grid: (M/128, Nout/64). Block: 128 threads = 4 waves; wave -> 32x64 tile.
// ---------------------------------------------------------------------------
__global__ __launch_bounds__(128) void wmma_gemm_k(
    const float* __restrict__ A, int D2v, int d12, long sB, long s1, long s2,
    const float* __restrict__ W, const float* __restrict__ bias,
    float* __restrict__ out, int K, int Nout) {
  __shared__ _Float16 bsm[2][2048];   // double-buffered [64 n][32 k]
  int wave = threadIdx.x >> 5, lane = threadIdx.x & 31;
  int nl = lane & 15, hh = lane >> 4;
  int m0 = blockIdx.x * 128 + wave * 32;
  int n0 = blockIdx.y * 64;
  const float* arow[2];
#pragma unroll
  for (int g = 0; g < 2; ++g) {
    int m = m0 + g * 16 + nl;
    int b = m / d12; int rr = m - b * d12;
    int i1 = rr / D2v; int i2 = rr - i1 * D2v;
    arow[g] = A + (long)b * sB + (long)i1 * s1 + (long)i2 * s2;
  }
  v8f acc[2][4] = {};

  // packed staging: 2 k-values per thread-iter -> v_cvt_pk_f16_f32 + ds_store_b32
  auto stage = [&](int buf, int k0) {
#pragma unroll
    for (int it = 0; it < 8; ++it) {
      int i = threadIdx.x + it * 128;         // 0..1023
      int kp = i >> 6, nn = i & 63;           // kp: k-pair 0..15
      float lo = W[(k0 + 2 * kp) * Nout + n0 + nn];
      float hi = W[(k0 + 2 * kp + 1) * Nout + n0 + nn];
      v2h p; p[0] = (_Float16)lo; p[1] = (_Float16)hi;
      *(v2h*)&bsm[buf][nn * 32 + 2 * kp] = p;
    }
  };

  stage(0, 0);
  __syncthreads();
  int nk = K >> 5;
  for (int kt = 0; kt < nk; ++kt) {
    int k0 = kt << 5;
    if (kt + 1 < nk) stage((kt + 1) & 1, k0 + 32);
    if (kt + 2 < nk)                                   // gfx1250 prefetch path
      __builtin_prefetch(&W[(long)(k0 + 64 + (threadIdx.x >> 6)) * Nout +
                            n0 + (threadIdx.x & 63)], 0, 1);
    const _Float16* bb = bsm[kt & 1];
    v16h fa0 = load_a_frag(arow[0] + k0, hh);
    v16h fa1 = load_a_frag(arow[1] + k0, hh);
#pragma unroll
    for (int s = 0; s < 4; ++s) {
      v16h fb = load_b_frag(&bb[(s * 16 + nl) * 32], hh);
      acc[0][s] = __builtin_amdgcn_wmma_f32_16x16x32_f16(
          false, fa0, false, fb, (short)0, acc[0][s], false, false);
      acc[1][s] = __builtin_amdgcn_wmma_f32_16x16x32_f16(
          false, fa1, false, fb, (short)0, acc[1][s], false, false);
    }
    __syncthreads();
  }
#pragma unroll
  for (int g = 0; g < 2; ++g) {
#pragma unroll
    for (int s = 0; s < 4; ++s) {
      int n = n0 + s * 16 + nl;
      float bv = bias ? bias[n] : 0.f;
#pragma unroll
      for (int r = 0; r < 8; ++r) {
        int mo = m0 + g * 16 + hh * 8 + r;            // D: lane->col, vgpr->row
        out[(long)mo * Nout + n] = acc[g][s][r] + bv;
      }
    }
  }
}

// ---------------------------------------------------------------------------
// Implicit-GEMM 3x3 SAME conv over channels-last [b][f][t][c] activations,
// OIHW weights. M = spatial (f,t), N = cout, K = 9*cin, k-step = 32 cin of
// one tap (72 steps). Optional fused bias + g_diff*temb + SiLU (conv1).
// Grid: (16384/128, 256/64, B). Block 128 = 4 waves; wave -> 32x64 tile.
// ---------------------------------------------------------------------------
__global__ __launch_bounds__(128) void wmma_conv3x3_k(
    const float* __restrict__ in, const float* __restrict__ w,
    const float* __restrict__ bias, const float* __restrict__ temb,
    const float* __restrict__ gdiff, float* __restrict__ out, int do_silu) {
  __shared__ _Float16 bsm[2][2048];
  int b = blockIdx.z;
  int wave = threadIdx.x >> 5, lane = threadIdx.x & 31;
  int nl = lane & 15, hh = lane >> 4;
  int m0 = blockIdx.x * 128 + wave * 32;
  int n0 = blockIdx.y * 64;
  int fr[2], tr[2];
#pragma unroll
  for (int g = 0; g < 2; ++g) {
    int m = m0 + g * 16 + nl;
    fr[g] = m >> 8; tr[g] = m & 255;
  }
  v8f acc[2][4] = {};

  auto stage = [&](int buf, int kt) {
    int tap = kt >> 3, c0 = (kt & 7) << 5;
#pragma unroll
    for (int it = 0; it < 8; ++it) {
      int i = threadIdx.x + it * 128;         // 0..1023
      int kp = i >> 6, nn = i & 63;           // k-pair: ci pair (stride 9 in w)
      long wb = ((long)(n0 + nn) * 256 + (c0 + 2 * kp)) * 9 + tap;
      float lo = w[wb];
      float hi = w[wb + 9];
      v2h p; p[0] = (_Float16)lo; p[1] = (_Float16)hi;
      *(v2h*)&bsm[buf][nn * 32 + 2 * kp] = p;
    }
  };

  stage(0, 0);
  __syncthreads();
  for (int kt = 0; kt < 72; ++kt) {
    if (kt + 1 < 72) stage((kt + 1) & 1, kt + 1);
    int tap = kt >> 3, c0 = (kt & 7) << 5;
    int dy = tap / 3 - 1, dx = tap % 3 - 1;
    const _Float16* bb = bsm[kt & 1];
    v16h fa[2];
#pragma unroll
    for (int g = 0; g < 2; ++g) {
      int fy = fr[g] + dy, tx = tr[g] + dx;
      bool valid = ((unsigned)fy < 64u) && ((unsigned)tx < 256u);
      fa[g] = v16h{};
      if (valid)                                       // zero-pad at borders
        fa[g] = load_a_frag(
            in + (((long)(b * 64 + fy) * 256 + tx) << 8) + c0, hh);
    }
#pragma unroll
    for (int s = 0; s < 4; ++s) {
      v16h fb = load_b_frag(&bb[(s * 16 + nl) * 32], hh);
      acc[0][s] = __builtin_amdgcn_wmma_f32_16x16x32_f16(
          false, fa[0], false, fb, (short)0, acc[0][s], false, false);
      acc[1][s] = __builtin_amdgcn_wmma_f32_16x16x32_f16(
          false, fa[1], false, fb, (short)0, acc[1][s], false, false);
    }
    __syncthreads();
  }
  float gd = gdiff ? gdiff[0] : 0.f;
#pragma unroll
  for (int g = 0; g < 2; ++g) {
#pragma unroll
    for (int s = 0; s < 4; ++s) {
      int n = n0 + s * 16 + nl;
      float bv = bias ? bias[n] : 0.f;
      float tv = temb ? gd * temb[(b << 8) + n] : 0.f;
#pragma unroll
      for (int r = 0; r < 8; ++r) {
        int mo = m0 + g * 16 + hh * 8 + r;
        int fo = mo >> 8, to = mo & 255;
        float v = acc[g][s][r] + bv + tv;
        if (do_silu) v = v / (1.f + __expf(-v));
        out[(((long)(b * 64 + fo) * 256 + to) << 8) + n] = v;
      }
    }
  }
}

// ---------------------------------------------------------------------------
// Layer2 attention over frequency (seq 64, Dh 32). Block=(bt,h), 64 threads.
// ---------------------------------------------------------------------------
__global__ void attn2_k(const float* __restrict__ qkv, float* __restrict__ y) {
  int bt = blockIdx.x, h = blockIdx.y;
  int L = threadIdx.x;                 // query index 0..63
  __shared__ float Ks[64 * 32];
  __shared__ float Vs[64 * 32];
  long base = (long)bt * 64 * 768 + h * 32;
  for (int i = L; i < 2048; i += 64) {
    int l = i >> 5, d = i & 31;
    Ks[i] = qkv[base + (long)l * 768 + 256 + d];
    Vs[i] = qkv[base + (long)l * 768 + 512 + d];
  }
  __syncthreads();
  float q[32];
#pragma unroll
  for (int d = 0; d < 32; ++d) q[d] = qkv[base + (long)L * 768 + d];
  float sc[64];
  float mx = -1e30f;
  for (int l = 0; l < 64; ++l) {
    float s = 0.f;
#pragma unroll
    for (int d = 0; d < 32; ++d) s += q[d] * Ks[l * 32 + d];
    s *= SCALE_;
    sc[l] = s; mx = fmaxf(mx, s);
  }
  float den = 0.f;
  for (int l = 0; l < 64; ++l) { sc[l] = __expf(sc[l] - mx); den += sc[l]; }
  float inv = 1.f / den;
  float acc[32];
#pragma unroll
  for (int d = 0; d < 32; ++d) acc[d] = 0.f;
  for (int l = 0; l < 64; ++l) {
    float p = sc[l] * inv;
#pragma unroll
    for (int d = 0; d < 32; ++d) acc[d] += p * Vs[l * 32 + d];
  }
  long ob = ((long)bt * 64 + L) * 256 + h * 32;
#pragma unroll
  for (int d = 0; d < 32; ++d) y[ob + d] = acc[d];
}

// per-row GN (eps 1e-5, 32 groups of 8) and gated residual into xf (in place)
__global__ void rowgn_add_k(const float* __restrict__ yin, float* __restrict__ xf,
                            const float* __restrict__ gate) {
  long row = blockIdx.x;
  int c = threadIdx.x;
  __shared__ float s1[32];
  __shared__ float s2[32];
  if (c < 32) { s1[c] = 0.f; s2[c] = 0.f; }
  __syncthreads();
  float v = yin[row * 256 + c];
  atomicAdd(&s1[c >> 3], v);
  atomicAdd(&s2[c >> 3], v * v);
  __syncthreads();
  float m = s1[c >> 3] * 0.125f;
  float var = s2[c >> 3] * 0.125f - m * m;
  xf[row * 256 + c] += gate[0] * (v - m) * rsqrtf(var + 1e-5f);
}

// per-row GN scaled by gate (layer3 small [256, C] tensor)
__global__ void rowgn_scale_k(const float* __restrict__ yin, float* __restrict__ out,
                              const float* __restrict__ gate) {
  long row = blockIdx.x;
  int c = threadIdx.x;
  __shared__ float s1[32];
  __shared__ float s2[32];
  if (c < 32) { s1[c] = 0.f; s2[c] = 0.f; }
  __syncthreads();
  float v = yin[row * 256 + c];
  atomicAdd(&s1[c >> 3], v);
  atomicAdd(&s2[c >> 3], v * v);
  __syncthreads();
  float m = s1[c >> 3] * 0.125f;
  float var = s2[c >> 3] * 0.125f - m * m;
  out[row * 256 + c] = gate[0] * (v - m) * rsqrtf(var + 1e-5f);
}

// ---------------------------------------------------------------------------
// Layer3: mean-pooled-query attention over time (T=256). Block=(bq,h), 256 thr.
// ---------------------------------------------------------------------------
__global__ void attn3_k(const float* __restrict__ qkv, float* __restrict__ y3) {
  int bq = blockIdx.x, h = blockIdx.y;
  int tid = threadIdx.x;               // key/time index 0..255
  __shared__ float qm[32];
  __shared__ float red[256];
  __shared__ float yacc[32];
  long base = (long)bq * 256 * 768 + h * 32;
  if (tid < 32) {
    float s = 0.f;
    for (int l = 0; l < 256; ++l) s += qkv[base + (long)l * 768 + tid];
    qm[tid] = s * (1.f / 256.f);
    yacc[tid] = 0.f;
  }
  __syncthreads();
  float s = 0.f;
#pragma unroll
  for (int d = 0; d < 32; ++d) s += qm[d] * qkv[base + (long)tid * 768 + 256 + d];
  s *= SCALE_;
  red[tid] = s;
  __syncthreads();
  for (int off = 128; off > 0; off >>= 1) {
    if (tid < off) red[tid] = fmaxf(red[tid], red[tid + off]);
    __syncthreads();
  }
  float mx = red[0];
  __syncthreads();
  float e = __expf(s - mx);
  red[tid] = e;
  __syncthreads();
  for (int off = 128; off > 0; off >>= 1) {
    if (tid < off) red[tid] += red[tid + off];
    __syncthreads();
  }
  float p = e / red[0];
#pragma unroll
  for (int d = 0; d < 32; ++d)
    atomicAdd(&yacc[d], p * qkv[base + (long)tid * 768 + 512 + d]);
  __syncthreads();
  if (tid < 32) y3[(long)bq * 256 + h * 32 + tid] = yacc[tid];
}

// x2_nchw = xf_cl2 + broadcast(y3g)   (y3g already gated)
__global__ void x2_build_k(const float* __restrict__ xf, const float* __restrict__ y3g,
                           float* __restrict__ x2) {
  int idx = blockIdx.x * 256 + threadIdx.x;   // NCHW flat
  int t = idx & 255, f = (idx >> 8) & 63, c = (idx >> 14) & 255, b = idx >> 22;
  float xv = xf[(((long)(b * 256 + t) * 64 + f) << 8) + c];
  x2[idx] = xv + y3g[((b * 64 + f) << 8) + c];
}

// ---------------------------------------------------------------------------
// Layer4 LSA: windows of 16 along T. Block=(bf, window), 128 thr = 16q x 8h.
// ---------------------------------------------------------------------------
__global__ void lsa_k(const float* __restrict__ qkv, float* __restrict__ o) {
  int n = blockIdx.x, w = blockIdx.y;
  int tid = threadIdx.x;
  int q = tid & 15, h = tid >> 4;
  __shared__ float Ks[16 * 256];
  __shared__ float Vs[16 * 256];
  long base = ((long)n * 256 + w * 16) * 768;
  for (int i = tid; i < 4096; i += 128) {
    int row = i >> 8, c = i & 255;
    Ks[i] = qkv[base + (long)row * 768 + 256 + c];
    Vs[i] = qkv[base + (long)row * 768 + 512 + c];
  }
  __syncthreads();
  float qv[32];
#pragma unroll
  for (int d = 0; d < 32; ++d) qv[d] = qkv[base + (long)q * 768 + h * 32 + d];
  float sc[16];
  float mx = -1e30f;
#pragma unroll
  for (int k = 0; k < 16; ++k) {
    float s = 0.f;
#pragma unroll
    for (int d = 0; d < 32; ++d) s += qv[d] * Ks[k * 256 + h * 32 + d];
    s *= SCALE_;
    sc[k] = s; mx = fmaxf(mx, s);
  }
  float den = 0.f;
#pragma unroll
  for (int k = 0; k < 16; ++k) { sc[k] = __expf(sc[k] - mx); den += sc[k]; }
  float inv = 1.f / den;
  float acc[32];
#pragma unroll
  for (int d = 0; d < 32; ++d) acc[d] = 0.f;
#pragma unroll
  for (int k = 0; k < 16; ++k) {
    float p = sc[k] * inv;
#pragma unroll
    for (int d = 0; d < 32; ++d) acc[d] += p * Vs[k * 256 + h * 32 + d];
  }
  long ob = ((long)n * 256 + w * 16 + q) * 256 + h * 32;
#pragma unroll
  for (int d = 0; d < 32; ++d) o[ob + d] = acc[d];
}

// out_nchw = x2_nchw + o2_rows(b,f,t)
__global__ void final_add_k(const float* __restrict__ x2, const float* __restrict__ o2,
                            float* __restrict__ out) {
  int idx = blockIdx.x * 256 + threadIdx.x;
  int t = idx & 255, f = (idx >> 8) & 63, c = (idx >> 14) & 255, b = idx >> 22;
  out[idx] = x2[idx] + o2[(((long)(b * 64 + f) * 256 + t) << 8) + c];
}

// ---------------------------------------------------------------------------
extern "C" void kernel_launch(void* const* d_in, const int* in_sizes, int n_in,
                              void* d_out, int out_size, void* d_ws, size_t ws_size,
                              hipStream_t stream) {
  const float* x       = (const float*)d_in[0];
  const float* temb    = (const float*)d_in[1];
  const float* conv1_w = (const float*)d_in[2];
  const float* conv1_b = (const float*)d_in[3];
  const float* conv2_w = (const float*)d_in[4];
  const float* g_diff  = (const float*)d_in[5];
  const float* g_res   = (const float*)d_in[6];
  const float* g2      = (const float*)d_in[7];
  const float* g3      = (const float*)d_in[8];
  const float* qkv2_w  = (const float*)d_in[9];
  const float* qkv2_b  = (const float*)d_in[10];
  const float* out2_w  = (const float*)d_in[11];
  const float* qkv3_w  = (const float*)d_in[12];
  const float* qkv3_b  = (const float*)d_in[13];
  const float* out3_w  = (const float*)d_in[14];
  const float* lsa_qkv_w = (const float*)d_in[15];
  const float* lsa_out_w = (const float*)d_in[16];
  float* out = (float*)d_out;

  const long N = (long)B_ * C_ * F_ * T_;        // 16,777,216
  float* ws = (float*)d_ws;
  float* W0    = ws;                             // g0/g1, y2, hl, o_attn
  float* W1    = ws + N;                         // h1/h2, y2o, o2
  float* QKV   = ws + 2 * N;                     // 3N
  float* XF    = ws + 5 * N;                     // xf (cl2 [b][t][f][c])
  float* X2    = ws + 6 * N;                     // x2 (NCHW)
  float* STATS = ws + 7 * N;                     // 256 floats
  float* Y3    = STATS + 256;                    // [256,256]
  float* Y3O   = Y3 + 65536;
  float* Y3G   = Y3O + 65536;

  const long sCFT = (long)C_ * F_ * T_;          // NCHW b-stride
  const long sFT  = (long)F_ * T_;               // NCHW c-stride
  const long sFTC = (long)F_ * T_ * C_;          // cl b-stride
  const long sTC  = (long)T_ * C_;               // cl1 f-stride
  const dim3 eg(65536);                          // elementwise grids (N/256)
  const dim3 cgrid(128, 4, 4);                   // conv: M/128, N/64, B
  const dim3 gq(512, 12);                        // 65536x768 GEMM
  const dim3 go(512, 4);                         // 65536x256 GEMM

  // ---- Residual block ----
  gn_reduce_k<<<128, 256, 0, stream>>>(x, STATS, sCFT, sFT, (long)T_, 1L, 1e-6f);
  gn_apply_nchw_to_cl1<<<eg, 256, 0, stream>>>(x, STATS, W0);
  wmma_conv3x3_k<<<cgrid, 128, 0, stream>>>(W0, conv1_w, conv1_b, temb, g_diff, W1, 1);
  gn_reduce_k<<<128, 256, 0, stream>>>(W1, STATS, sFTC, 1L, sTC, (long)C_, 1e-6f);
  gn_apply_cl1<<<eg, 256, 0, stream>>>(W1, STATS, W0);
  wmma_conv3x3_k<<<cgrid, 128, 0, stream>>>(W0, conv2_w, nullptr, nullptr, nullptr, W1, 0);
  gn_reduce_k<<<128, 256, 0, stream>>>(W1, STATS, sFTC, 1L, sTC, (long)C_, 1e-6f);
  gn_apply_res_to_cl2<<<eg, 256, 0, stream>>>(x, W1, STATS, g_res, XF);

  // ---- Layer2: frequency attention (rows (b,t,f) = identity on XF) ----
  wmma_gemm_k<<<gq, 128, 0, stream>>>(XF, 64, 16384, sFTC, (long)F_ * C_, (long)C_,
                                      qkv2_w, qkv2_b, QKV, 256, 768);
  attn2_k<<<dim3(1024, 8), 64, 0, stream>>>(QKV, W0);
  wmma_gemm_k<<<go, 128, 0, stream>>>(W0, 64, 16384, sFTC, (long)F_ * C_, (long)C_,
                                      out2_w, nullptr, W1, 256, 256);
  rowgn_add_k<<<65536, 256, 0, stream>>>(W1, XF, g2);

  // ---- Layer3: global time attention (rows (b,f,t) gathered from cl2) ----
  wmma_gemm_k<<<gq, 128, 0, stream>>>(XF, 256, 16384, sFTC, (long)C_, (long)F_ * C_,
                                      qkv3_w, qkv3_b, QKV, 256, 768);
  attn3_k<<<dim3(256, 8), 256, 0, stream>>>(QKV, Y3);
  wmma_gemm_k<<<dim3(2, 4), 128, 0, stream>>>(Y3, 1, 1, (long)C_, 0L, 0L,
                                              out3_w, nullptr, Y3O, 256, 256);
  rowgn_scale_k<<<256, 256, 0, stream>>>(Y3O, Y3G, g3);
  x2_build_k<<<eg, 256, 0, stream>>>(XF, Y3G, X2);

  // ---- Layer4: LSA ----
  gn_reduce_k<<<128, 256, 0, stream>>>(X2, STATS, sCFT, sFT, (long)T_, 1L, 1e-5f);
  gn_apply_nchw_to_cl1<<<eg, 256, 0, stream>>>(X2, STATS, W0);   // hl rows (b,f,t)
  wmma_gemm_k<<<gq, 128, 0, stream>>>(W0, 256, 16384, sFTC, sTC, (long)C_,
                                      lsa_qkv_w, nullptr, QKV, 256, 768);
  lsa_k<<<dim3(256, 16), 128, 0, stream>>>(QKV, W0);             // o_attn reuses W0
  wmma_gemm_k<<<go, 128, 0, stream>>>(W0, 256, 16384, sFTC, sTC, (long)C_,
                                      lsa_out_w, nullptr, W1, 256, 256);
  final_add_k<<<eg, 256, 0, stream>>>(X2, W1, out);

  (void)in_sizes; (void)n_in; (void)out_size; (void)ws_size;
}
